// SelfAttention_59322088292466
// MI455X (gfx1250) — matrix operand order repaired and checked by reading, not compiled, not run on hardware
//
#include <hip/hip_runtime.h>
#include <math.h>

// ---------------------------------------------------------------------------
// Types
// ---------------------------------------------------------------------------
typedef __attribute__((ext_vector_type(8)))  __bf16 v8bf;
typedef __attribute__((ext_vector_type(16))) __bf16 v16bf;
typedef __attribute__((ext_vector_type(8)))  float  v8f;

constexpr int cB  = 2;
constexpr int cS  = 2048;
constexpr int cDM = 2048;
constexpr int cH  = 16;
constexpr int cDH = 128;
constexpr int cBS = cB * cS;                 // 4096 total rows
constexpr float cQScale = 0.08838834764831845f;  // 1/sqrt(128)

enum { MODE_QK = 0, MODE_VT = 1, MODE_F32 = 2 };

// ---------------------------------------------------------------------------
// Helpers
// ---------------------------------------------------------------------------
__device__ __forceinline__ unsigned short f2bf_bits(float f) {
  unsigned u = __builtin_bit_cast(unsigned, f);
  u += 0x7fffu + ((u >> 16) & 1u);           // round-to-nearest-even
  return (unsigned short)(u >> 16);
}

// A fragment (16-bit, 16x32, M x K): lane holds row r, K = kb..kb+15 contiguous
// where kb = k0 + (lane>=16 ? 16 : 0) is passed in as `kbase`.
__device__ __forceinline__ v16bf load_a_frag(const unsigned short* __restrict__ base,
                                             int ld, int row, int kbase) {
  const unsigned short* p = base + (size_t)row * ld + kbase;
  v8bf lo = *(const v8bf*)(p);
  v8bf hi = *(const v8bf*)(p + 8);
  return __builtin_shufflevector(lo, hi, 0,1,2,3,4,5,6,7,8,9,10,11,12,13,14,15);
}

// B fragment (16-bit, 32x16, K x N), column-major storage Bt[n][k]:
// lane holds column n, K = kb..kb+15 contiguous, kb = k0 + (lane>=16 ? 16 : 0).
__device__ __forceinline__ v16bf load_b_frag(const unsigned short* __restrict__ base,
                                             int ld, int col, int kbase) {
  return *(const v16bf*)(base + (size_t)col * ld + kbase);
}

__device__ __forceinline__ float red_max16(float v) {
#pragma unroll
  for (int m = 1; m < 16; m <<= 1) v = fmaxf(v, __shfl_xor(v, m, 16));
  return v;
}
__device__ __forceinline__ float red_sum16(float v) {
#pragma unroll
  for (int m = 1; m < 16; m <<= 1) v += __shfl_xor(v, m, 16);
  return v;
}

// ---------------------------------------------------------------------------
// Prep kernels
// ---------------------------------------------------------------------------
__global__ void cast_f32_to_bf16(const float* __restrict__ in,
                                 unsigned short* __restrict__ out, size_t n) {
  size_t i = (size_t)blockIdx.x * blockDim.x + threadIdx.x;
  size_t stride = (size_t)gridDim.x * blockDim.x;
  for (; i < n; i += stride) out[i] = f2bf_bits(in[i]);
}

// W[K][N] f32 row-major  ->  Wt[N][K] bf16 row-major (i.e. column-major W)
__global__ __launch_bounds__(256)
void transpose_cast_bf16(const float* __restrict__ W,
                         unsigned short* __restrict__ Wt, int K, int N) {
  __shared__ float tile[32][33];
  int bx = blockIdx.x * 32;   // n tile
  int by = blockIdx.y * 32;   // k tile
  int tx = threadIdx.x;       // 0..31
  int ty = threadIdx.y;       // 0..7
#pragma unroll
  for (int i = 0; i < 32; i += 8)
    tile[ty + i][tx] = W[(size_t)(by + ty + i) * N + bx + tx];
  __syncthreads();
#pragma unroll
  for (int i = 0; i < 32; i += 8)
    Wt[(size_t)(bx + ty + i) * K + by + tx] = f2bf_bits(tile[tx][ty + i]);
}

// ---------------------------------------------------------------------------
// WMMA GEMM: C[M,N] = A[M,K] @ Bt[N,K]^T + bias, with mode-specific epilogue.
// One wave computes a 32x64 tile. The K loop is unrolled 2x so the two
// fragment buffers alternate roles by NAME (no register-copy rotation) while
// loads for step k+1 overlap the 8-WMMA chain of step k.
// Requires K % 64 == 0 (true for all uses here: K = 2048).
// ---------------------------------------------------------------------------
__global__ __launch_bounds__(128)
void gemm_bf16_wmma(const unsigned short* __restrict__ A,
                    const unsigned short* __restrict__ Bt,
                    const float* __restrict__ bias,
                    void* __restrict__ outp,
                    int M, int N, int K, int mode, float scale) {
  const int lane = threadIdx.x & 31;
  const int wave = threadIdx.x >> 5;
  const int half = lane >> 4;
  const int r16  = lane & 15;
  const int m0 = blockIdx.y * 32;
  const int n0 = (blockIdx.x * 4 + wave) * 64;

  const int ka = half * 16;   // A-frag K pattern base offset
  const int kb = half * 16;   // B-frag K pattern base offset

  v8f acc[2][4];
#pragma unroll
  for (int mi = 0; mi < 2; mi++)
#pragma unroll
    for (int j = 0; j < 4; j++)
#pragma unroll
      for (int i = 0; i < 8; i++) acc[mi][j][i] = 0.0f;

  auto load_step = [&](v16bf& A0, v16bf& A1, v16bf (&Bf)[4], int k0) {
    A0 = load_a_frag(A, K, m0 + r16,      k0 + ka);
    A1 = load_a_frag(A, K, m0 + 16 + r16, k0 + ka);
#pragma unroll
    for (int j = 0; j < 4; j++) Bf[j] = load_b_frag(Bt, K, n0 + j * 16 + r16, k0 + kb);
  };
  auto mma_step = [&](const v16bf& A0, const v16bf& A1, const v16bf (&Bf)[4]) {
#pragma unroll
    for (int j = 0; j < 4; j++) {
      acc[0][j] = __builtin_amdgcn_wmma_f32_16x16x32_bf16(false, A0, false, Bf[j],
                                                          (short)0, acc[0][j], false, false);
      acc[1][j] = __builtin_amdgcn_wmma_f32_16x16x32_bf16(false, A1, false, Bf[j],
                                                          (short)0, acc[1][j], false, false);
    }
  };

  v16bf a0A, a1A, bA[4];
  v16bf a0B, a1B, bB[4];

  load_step(a0A, a1A, bA, 0);
  int k0 = 0;
  for (; k0 + 64 < K; k0 += 64) {
    load_step(a0B, a1B, bB, k0 + 32);   // overlap with mma(bufA)
    mma_step(a0A, a1A, bA);
    load_step(a0A, a1A, bA, k0 + 64);   // overlap with mma(bufB)
    mma_step(a0B, a1B, bB);
  }
  // tail: last two K-steps
  load_step(a0B, a1B, bB, k0 + 32);
  mma_step(a0A, a1A, bA);
  mma_step(a0B, a1B, bB);

  // Epilogue: element (i) of a D fragment is row (i + half*8), col r16.
#pragma unroll
  for (int mi = 0; mi < 2; mi++) {
#pragma unroll
    for (int j = 0; j < 4; j++) {
#pragma unroll
      for (int i = 0; i < 8; i++) {
        int gm = m0 + mi * 16 + i + half * 8;
        int gn = n0 + j * 16 + r16;
        float v = (acc[mi][j][i] + bias[gn]) * scale;
        if (mode == MODE_F32) {
          ((float*)outp)[(size_t)gm * N + gn] = v;
        } else {
          int b = gm >> 11;          // gm / cS
          int s = gm & (cS - 1);
          int h = gn >> 7;           // gn / cDH
          int d = gn & (cDH - 1);
          size_t off;
          if (mode == MODE_QK)
            off = (((size_t)b * cH + h) * cS + s) * cDH + d;   // [b,h,s,dh]
          else
            off = (((size_t)b * cH + h) * cDH + d) * cS + s;   // [b,h,dh,s]
          ((unsigned short*)outp)[off] = f2bf_bits(v);
        }
      }
    }
  }
}

// ---------------------------------------------------------------------------
// Flash attention: one wave per (b, h, 16-row q tile). Causal + pad mask.
// Q is pre-scaled by 1/sqrt(DH). V fragments are fetched early so their
// latency hides under the softmax VALU work. Output bf16 to attn[b,s,dm].
// ---------------------------------------------------------------------------
__global__ __launch_bounds__(32)
void attn_fwd(const unsigned short* __restrict__ Q,
              const unsigned short* __restrict__ Kmat,
              const unsigned short* __restrict__ Vt,
              const unsigned char* __restrict__ pad,
              unsigned short* __restrict__ attn) {
  __shared__ unsigned short pbuf[16 * 32];   // P tile bounce for layout transpose

  const int lane = threadIdx.x & 31;
  const int half = lane >> 4;
  const int r16  = lane & 15;
  const int q0 = blockIdx.x * 16;
  const int h  = blockIdx.y;
  const int b  = blockIdx.z;

  const unsigned short* Qb = Q    + (((size_t)b * cH + h) * cS) * cDH;
  const unsigned short* Kb = Kmat + (((size_t)b * cH + h) * cS) * cDH;
  const unsigned short* Vb = Vt   + (((size_t)b * cH + h) * cDH) * cS;
  const unsigned char*  pb = pad  + (size_t)b * cS;

  // Q fragments for the 4 K-chunks of DH=128
  v16bf qf[4];
#pragma unroll
  for (int j = 0; j < 4; j++)
    qf[j] = load_a_frag(Qb, cDH, q0 + r16, j * 32 + half * 16);

  v8f o[8];
  float mrun[8], lrun[8];
#pragma unroll
  for (int t = 0; t < 8; t++)
#pragma unroll
    for (int i = 0; i < 8; i++) o[t][i] = 0.0f;
#pragma unroll
  for (int i = 0; i < 8; i++) { mrun[i] = -INFINITY; lrun[i] = 0.0f; }

  const int kend = q0 + 16;                  // causal: kv tiles with k0 < q0+16
  for (int k0 = 0; k0 < kend; k0 += 32) {
    // ---- scores: S = Q @ K^T, two 16x16 n-tiles (kv cols k0..15, 16..31) ----
    v8f s0, s1;
#pragma unroll
    for (int i = 0; i < 8; i++) { s0[i] = 0.0f; s1[i] = 0.0f; }
#pragma unroll
    for (int j = 0; j < 4; j++) {
      v16bf kb0 = load_b_frag(Kb, cDH, k0 + r16,      j * 32 + half * 16);
      v16bf kb1 = load_b_frag(Kb, cDH, k0 + 16 + r16, j * 32 + half * 16);
      s0 = __builtin_amdgcn_wmma_f32_16x16x32_bf16(false, qf[j], false, kb0,
                                                   (short)0, s0, false, false);
      s1 = __builtin_amdgcn_wmma_f32_16x16x32_bf16(false, qf[j], false, kb1,
                                                   (short)0, s1, false, false);
    }

    // ---- prefetch V fragments now: independent of softmax, latency hides
    //      under the ~100 VALU ops below ----
    v16bf vb[8];
#pragma unroll
    for (int t = 0; t < 8; t++)
      vb[t] = load_b_frag(Vb, cS, t * 16 + r16, k0 + half * 16);

    // ---- mask + online softmax ----
    const int kv0 = k0 + r16;
    const int kv1 = k0 + 16 + r16;
    const bool p0 = pb[kv0] != 0;
    const bool p1 = pb[kv1] != 0;

    float mx[8];
#pragma unroll
    for (int i = 0; i < 8; i++) {
      const int qrow = q0 + i + half * 8;
      float v0 = (p0 && kv0 <= qrow) ? s0[i] : -INFINITY;
      float v1 = (p1 && kv1 <= qrow) ? s1[i] : -INFINITY;
      s0[i] = v0; s1[i] = v1;
      mx[i] = red_max16(fmaxf(v0, v1));
    }

    float corr[8];
#pragma unroll
    for (int i = 0; i < 8; i++) {
      float mn = fmaxf(mrun[i], mx[i]);
      corr[i]  = __expf(mrun[i] - mn);
      mrun[i]  = mn;
      float e0 = __expf(s0[i] - mn);
      float e1 = __expf(s1[i] - mn);
      s0[i] = e0; s1[i] = e1;
      lrun[i] = lrun[i] * corr[i] + red_sum16(e0 + e1);
    }
#pragma unroll
    for (int t = 0; t < 8; t++)
#pragma unroll
      for (int i = 0; i < 8; i++) o[t][i] *= corr[i];

    // ---- P (D-frag layout) -> bf16 A-frag layout via LDS ----
    __syncthreads();
#pragma unroll
    for (int i = 0; i < 8; i++) {
      pbuf[(i + half * 8) * 32 + r16]      = f2bf_bits(s0[i]);
      pbuf[(i + half * 8) * 32 + 16 + r16] = f2bf_bits(s1[i]);
    }
    __syncthreads();
    v16bf pf;
    {
      const unsigned short* p = pbuf + r16 * 32 + half * 16;
      v8bf lo = *(const v8bf*)(p);
      v8bf hi = *(const v8bf*)(p + 8);
      pf = __builtin_shufflevector(lo, hi, 0,1,2,3,4,5,6,7,8,9,10,11,12,13,14,15);
    }

    // ---- O += P @ V ----
#pragma unroll
    for (int t = 0; t < 8; t++) {
      o[t] = __builtin_amdgcn_wmma_f32_16x16x32_bf16(false, pf, false, vb[t],
                                                     (short)0, o[t], false, false);
    }
  }

  // ---- epilogue: normalize and store bf16 to attn[b, s, h*DH + d] ----
  float inv[8];
#pragma unroll
  for (int i = 0; i < 8; i++) inv[i] = 1.0f / lrun[i];
#pragma unroll
  for (int t = 0; t < 8; t++) {
#pragma unroll
    for (int i = 0; i < 8; i++) {
      const int s = q0 + i + half * 8;
      const int d = h * cDH + t * 16 + r16;
      attn[((size_t)b * cS + s) * cDM + d] = f2bf_bits(o[t][i] * inv[i]);
    }
  }
}

// ---------------------------------------------------------------------------
// Host launcher
// ---------------------------------------------------------------------------
extern "C" void kernel_launch(void* const* d_in, const int* in_sizes, int n_in,
                              void* d_out, int out_size, void* d_ws, size_t ws_size,
                              hipStream_t stream) {
  const float*         x     = (const float*)d_in[0];
  const unsigned char* amask = (const unsigned char*)d_in[1];
  const float* wq = (const float*)d_in[2];
  const float* bq = (const float*)d_in[3];
  const float* wk = (const float*)d_in[4];
  const float* bk = (const float*)d_in[5];
  const float* wv = (const float*)d_in[6];
  const float* bv = (const float*)d_in[7];
  const float* wo = (const float*)d_in[8];
  const float* bo = (const float*)d_in[9];

  // Workspace layout (bf16 = ushort), total ~112 MB
  const size_t nXW = (size_t)cBS * cDM;      // 8,388,608
  const size_t nW  = (size_t)cDM * cDM;      // 4,194,304
  char* ws = (char*)d_ws;
  unsigned short* x_bf  = (unsigned short*)(ws);                 size_t off = nXW * 2;
  unsigned short* wqT   = (unsigned short*)(ws + off);           off += nW * 2;
  unsigned short* wkT   = (unsigned short*)(ws + off);           off += nW * 2;
  unsigned short* wvT   = (unsigned short*)(ws + off);           off += nW * 2;
  unsigned short* woT   = (unsigned short*)(ws + off);           off += nW * 2;
  unsigned short* Qb    = (unsigned short*)(ws + off);           off += nXW * 2;
  unsigned short* Kbuf  = (unsigned short*)(ws + off);           off += nXW * 2;
  unsigned short* Vt    = (unsigned short*)(ws + off);           off += nXW * 2;
  unsigned short* attnb = (unsigned short*)(ws + off);           off += nXW * 2;

  // 1) x -> bf16
  cast_f32_to_bf16<<<4096, 256, 0, stream>>>(x, x_bf, nXW);

  // 2) weights -> transposed bf16 (Wt[n][k])
  {
    dim3 tb(32, 8), tg(cDM / 32, cDM / 32);
    transpose_cast_bf16<<<tg, tb, 0, stream>>>(wq, wqT, cDM, cDM);
    transpose_cast_bf16<<<tg, tb, 0, stream>>>(wk, wkT, cDM, cDM);
    transpose_cast_bf16<<<tg, tb, 0, stream>>>(wv, wvT, cDM, cDM);
    transpose_cast_bf16<<<tg, tb, 0, stream>>>(wo, woT, cDM, cDM);
  }

  // 3) Q/K/V projections (Q pre-scaled by 1/sqrt(DH); V stored transposed)
  dim3 gb(128);
  dim3 gg(cDM / 256, cBS / 32);   // (8, 128)
  gemm_bf16_wmma<<<gg, gb, 0, stream>>>(x_bf, wqT, bq, Qb,   cBS, cDM, cDM, MODE_QK, cQScale);
  gemm_bf16_wmma<<<gg, gb, 0, stream>>>(x_bf, wkT, bk, Kbuf, cBS, cDM, cDM, MODE_QK, 1.0f);
  gemm_bf16_wmma<<<gg, gb, 0, stream>>>(x_bf, wvT, bv, Vt,   cBS, cDM, cDM, MODE_VT, 1.0f);

  // 4) flash attention
  attn_fwd<<<dim3(cS / 16, cH, cB), 32, 0, stream>>>(Qb, Kbuf, Vt, amask, attnb);

  // 5) output projection -> f32 d_out
  gemm_bf16_wmma<<<gg, gb, 0, stream>>>(attnb, woT, bo, d_out, cBS, cDM, cDM, MODE_F32, 1.0f);
}
